// SofaNetEllipse_38525856645310
// MI455X (gfx1250) — compile-verified, matrix operand order checked by reading, not run on hardware
//
#include <hip/hip_runtime.h>
#include <hip/hip_bf16.h>
#include <math.h>

typedef __attribute__((ext_vector_type(16))) _Float16 v16h;
typedef __attribute__((ext_vector_type(8)))  float    v8f;

#define N_AB  1024
#define NFULL 2049
#define HALF  1025
#define H     64

// ---- hardware transcendentals (gfx1250 V_TANH_F32 if available) ----
#if __has_builtin(__builtin_amdgcn_tanhf)
static __device__ __forceinline__ float fast_tanh(float x) { return __builtin_amdgcn_tanhf(x); }
#elif __has_builtin(__builtin_amdgcn_tanh_f32)
static __device__ __forceinline__ float fast_tanh(float x) { return __builtin_amdgcn_tanh_f32(x); }
#else
static __device__ __forceinline__ float fast_tanh(float x) { return tanhf(x); }
#endif

// One block per network n. 256 threads = 8 wave32.
// Each wave processes 16-column tiles of the 1025 angle samples via
// v_wmma_f32_16x16x32_f16 (K=64 -> 2 chained WMMAs), primal + JVP tangent.
__global__ __launch_bounds__(256) void sofa_ellipse_kernel(
    const float* __restrict__ alpha,  // (2049)
    const float* __restrict__ w0,     // (N_AB, H, 1)
    const float* __restrict__ w1,     // (N_AB, H, H)
    const float* __restrict__ w2,     // (N_AB, 2, H)
    const float* __restrict__ b0,     // (N_AB, H)
    const float* __restrict__ b1,     // (N_AB, H)
    const float* __restrict__ b2,     // (N_AB, 2)
    float* __restrict__ out)          // xp|yp|xpp|ypp, each (N_AB, NFULL)
{
    __shared__ _Float16 s_w1[H * H];          // w1 in f16 for WMMA A operand
    __shared__ float s_w0[H], s_b0[H], s_b1[H], s_w2[2 * H], s_b2[2];
    __shared__ float s_xp[HALF];              // xp first half (for mirror pass)

    const int n    = blockIdx.x;
    const int tid  = threadIdx.x;
    const int lane = tid & 31;
    const int wave = tid >> 5;

    // ---- cooperative weight staging ----
    for (int i = tid; i < H * H; i += 256)
        s_w1[i] = (_Float16)w1[(size_t)n * H * H + i];
    if (tid < H) {
        s_w0[tid] = w0[(size_t)n * H + tid];
        s_b0[tid] = b0[(size_t)n * H + tid];
        s_b1[tid] = b1[(size_t)n * H + tid];
    }
    if (tid < 2 * H) s_w2[tid] = w2[(size_t)n * 2 * H + tid];
    if (tid < 2)     s_b2[tid] = b2[(size_t)n * 2 + tid];
    __syncthreads();

    const int mrow = lane & 15;   // row (A) / column (B) index within tile
    const int grp  = lane >> 4;   // half-wave group

    // ---- load w1 into registers in WMMA 16-bit A layout (16x32 per tile) ----
    // half h of lane L holds K = h + 8*grp + 8*(h>=8)
    v16h A[4][2];
#pragma unroll
    for (int mt = 0; mt < 4; ++mt) {
#pragma unroll
        for (int kt = 0; kt < 2; ++kt) {
            v16h a;
#pragma unroll
            for (int h = 0; h < 16; ++h) {
                const int k = h + 8 * grp + ((h >= 8) ? 8 : 0) + 32 * kt;
                a[h] = s_w1[(16 * mt + mrow) * H + k];
            }
            A[mt][kt] = a;
        }
    }

    const float b2_0 = s_b2[0], b2_1 = s_b2[1];
    const size_t SZ   = (size_t)N_AB * NFULL;
    const size_t rowb = (size_t)n * NFULL;

    // ---- column tiles: 65 tiles of 16 (last tile has 1 valid column) ----
    for (int tile = wave; tile < 65; tile += 8) {
        const int c  = tile * 16 + mrow;       // this lane's column
        const int cc = (c > 1024) ? 1024 : c;  // clamp for the ragged tile
        const float x = alpha[cc];

        // B operands (layer-1 activations), same K mapping as A, col = mrow
        v16h Bp[2], Bt[2];
#pragma unroll
        for (int kt = 0; kt < 2; ++kt) {
            v16h bp, bt;
#pragma unroll
            for (int h = 0; h < 16; ++h) {
                const int k  = h + 8 * grp + ((h >= 8) ? 8 : 0) + 32 * kt;
                const float wk = s_w0[k];
                const float h0 = fast_tanh(fmaf(wk, x, s_b0[k]));  // V_TANH_F32
                const float d0 = (1.0f - h0 * h0) * wk;            // JVP of layer 1
                bp[h] = (_Float16)h0;
                bt[h] = (_Float16)d0;
            }
            Bp[kt] = bp;
            Bt[kt] = bt;
        }

        // layer-2 WMMA + layer-3 per-lane reduction
        float p0 = 0.f, p1 = 0.f, q0 = 0.f, q1 = 0.f;
#pragma unroll
        for (int mt = 0; mt < 4; ++mt) {
            v8f accp = {};
            v8f acct = {};
            accp = __builtin_amdgcn_wmma_f32_16x16x32_f16(false, A[mt][0], false, Bp[0], (short)0, accp, false, false);
            accp = __builtin_amdgcn_wmma_f32_16x16x32_f16(false, A[mt][1], false, Bp[1], (short)0, accp, false, false);
            acct = __builtin_amdgcn_wmma_f32_16x16x32_f16(false, A[mt][0], false, Bt[0], (short)0, acct, false, false);
            acct = __builtin_amdgcn_wmma_f32_16x16x32_f16(false, A[mt][1], false, Bt[1], (short)0, acct, false, false);
#pragma unroll
            for (int r = 0; r < 8; ++r) {
                const int row = 16 * mt + 8 * grp + r;  // C layout: VGPR r = row 8*grp+r
                const float h1 = fast_tanh(accp[r] + s_b1[row]);   // V_TANH_F32
                const float d1 = (1.0f - h1 * h1) * acct[r];
                p0 = fmaf(s_w2[row],     h1, p0);
                p1 = fmaf(s_w2[H + row], h1, p1);
                q0 = fmaf(s_w2[row],     d1, q0);
                q1 = fmaf(s_w2[H + row], d1, q1);
            }
        }
        // sum the two half-wave groups (rows 0-7 vs 8-15 of each M tile)
        p0 += __shfl_xor(p0, 16, 32);
        p1 += __shfl_xor(p1, 16, 32);
        q0 += __shfl_xor(q0, 16, 32);
        q1 += __shfl_xor(q1, 16, 32);

        const float out0 = p0 + b2_0;
        const float out1 = p1 + b2_1;
        const float av = out0 * out0;
        const float bv = out1 * out1;
        const float da = 2.0f * out0 * q0;
        const float db = 2.0f * out1 * q1;

        const float ca = __cosf(x);   // v_cos_f32; x in [0, pi/2]
        const float sa = __sinf(x);   // v_sin_f32
        const float xp  = av * (ca - 1.0f);
        const float yp  = bv * sa;
        const float xpp = fmaf(da, (ca - 1.0f), -av * sa);
        const float ypp = fmaf(db, sa, bv * ca);

        if (grp == 0 && c <= 1024) {
            const size_t base = rowb + c;
            out[base]          = xp;
            s_xp[c]            = xp;
            out[SZ + base]     = yp;
            out[2 * SZ + base] = xpp;
            out[3 * SZ + base] = ypp;
            if (c < 1024) {                       // mirror half (no xp dependency)
                const size_t mb = rowb + (2048 - c);
                out[SZ + mb]     = yp;
                out[2 * SZ + mb] = xpp;
                out[3 * SZ + mb] = -ypp;
            }
        }
    }

    // ---- xp mirror needs xp[:,1024] (cross-wave) -> barrier then LDS pass ----
    __syncthreads();
    const float xl = s_xp[1024];
    for (int c2 = tid; c2 < 1024; c2 += 256)
        out[rowb + (2048 - c2)] = 2.0f * xl - s_xp[c2];
}

extern "C" void kernel_launch(void* const* d_in, const int* in_sizes, int n_in,
                              void* d_out, int out_size, void* d_ws, size_t ws_size,
                              hipStream_t stream) {
    const float* alpha = (const float*)d_in[0];
    const float* w0    = (const float*)d_in[1];
    const float* w1    = (const float*)d_in[2];
    const float* w2    = (const float*)d_in[3];
    const float* b0    = (const float*)d_in[4];
    const float* b1    = (const float*)d_in[5];
    const float* b2    = (const float*)d_in[6];
    (void)in_sizes; (void)n_in; (void)out_size; (void)d_ws; (void)ws_size;

    sofa_ellipse_kernel<<<N_AB, 256, 0, stream>>>(alpha, w0, w1, w2, b0, b1, b2,
                                                  (float*)d_out);
}
